// WindowMultiHeadSelfAttention_14199161880713
// MI455X (gfx1250) — compile-verified, matrix operand order checked by reading, not run on hardware
//
#include <hip/hip_runtime.h>
#include <hip/hip_bf16.h>

// ---------------------------------------------------------------------------
// Swin-style window MHSA, fused per-window, bf16 WMMA on gfx1250 (MI455X).
//   x:     f32 [16][384][56][56]  (NCHW)
//   Wqkv:  f32 [384][1152], bqkv f32 [1152]
//   Wproj: f32 [384][384],  bproj f32 [384]
//   out:   f32 [16][384][56][56]
// 1024 windows (16 imgs x 8x8), each 49 tokens (padded to 64) x 384 dims.
// All intermediates live in ~271KB of the 320KB/WGP LDS -> only x-in / out
// touch HBM (~156MB total), weights (1.1MB bf16) stay L2-resident.
// ---------------------------------------------------------------------------

typedef __bf16 bf16;
typedef __attribute__((ext_vector_type(16))) __bf16 v16bf;
typedef __attribute__((ext_vector_type(8)))  __bf16 v8bf;
typedef __attribute__((ext_vector_type(8)))  float  v8f;

#define DIM       384
#define THREE_DIM 1152
#define HEADS     12
#define HD        32
#define N_TOK     49
#define NP        64        // padded token count (4 M-tiles of 16)
#define XS_STRIDE 400       // bf16 elements, multiple of 8 (16B rows)
#define QS_STRIDE 392
#define S_STRIDE  68        // f32 scores stride
#define P_STRIDE  72        // bf16 probs stride

// LDS element counts (bf16 unless noted)
#define XS_ELEMS   (NP * XS_STRIDE)           // 25600
#define QS_ELEMS   (NP * QS_STRIDE)           // 25088
#define VT_ELEMS   (HEADS * HD * NP)          // 24576
#define PS_ELEMS   (NP * P_STRIDE)            // 4608
#define SC_ELEMS   (NP * S_STRIDE)            // f32: 4352
#define SMEM_BYTES ((XS_ELEMS + 3*QS_ELEMS + VT_ELEMS + PS_ELEMS) * 2 + SC_ELEMS * 4 + NP * 4)

__device__ __forceinline__ v8f wmma_bf16(v16bf a, v16bf b, v8f c) {
  // D = A(16x32 bf16) x B(32x16 bf16) + C(16x16 f32)
  return __builtin_amdgcn_wmma_f32_16x16x32_bf16(
      /*neg_a=*/false, a, /*neg_b=*/false, b,
      /*c_mod=*/(short)0, c, /*reuse_a=*/false, /*reuse_b=*/false);
}

// Load a 16-bit A- or Bt-fragment from a row-major matrix.
// ISA layout: lanes 0-15 -> row = r0+lane, K = k0+{0..7, 16..23};
//             lanes 16-31 -> row = r0+lane-16, K = k0+{8..15, 24..31}.
// Two contiguous 16-byte loads per lane.
__device__ __forceinline__ v16bf load_frag(const bf16* __restrict__ base,
                                           int r0, int k0, int stride, int lane) {
  const int r  = r0 + (lane & 15);
  const int kb = k0 + ((lane >> 4) << 3);
  const bf16* p = base + r * stride + kb;
  v8bf lo = *(const v8bf*)(p);
  v8bf hi = *(const v8bf*)(p + 16);
  v16bf f;
#pragma unroll
  for (int i = 0; i < 8; ++i) { f[i] = lo[i]; f[i + 8] = hi[i]; }
  return f;
}

// ---------------------------------------------------------------------------
// Prep: transpose + convert weights to bf16 (stays resident in 192MB L2).
//   wqkvT[j][d] = Wqkv[d][j]   (1152 x 384)
//   wprojT[j][d] = Wproj[d][j] (384 x 384)
// ---------------------------------------------------------------------------
__global__ __launch_bounds__(256) void prep_weights(
    const float* __restrict__ wqkv, const float* __restrict__ wproj,
    bf16* __restrict__ wqkvT, bf16* __restrict__ wprojT) {
  int i = blockIdx.x * blockDim.x + threadIdx.x;
  if (i < DIM * THREE_DIM) {
    int j = i / DIM, d = i - j * DIM;
    wqkvT[i] = (bf16)wqkv[(size_t)d * THREE_DIM + j];
  }
  int k = i - DIM * THREE_DIM;
  if (k >= 0 && k < DIM * DIM) {
    int j = k / DIM, d = k - j * DIM;
    wprojT[k] = (bf16)wproj[(size_t)d * DIM + j];
  }
}

// ---------------------------------------------------------------------------
// Fused window attention: one workgroup (8 wave32) per window.
// ---------------------------------------------------------------------------
__global__ __launch_bounds__(256) void win_attn_fused(
    const float* __restrict__ x,
    const bf16*  __restrict__ wqkvT, const float* __restrict__ bqkv,
    const bf16*  __restrict__ wprojT, const float* __restrict__ bproj,
    float* __restrict__ out) {
  extern __shared__ __align__(16) char smem_raw[];
  bf16*  Xs   = (bf16*)smem_raw;            // [64][400] padded window input
  bf16*  Qs   = Xs  + XS_ELEMS;             // [64][392] Q, head-major cols
  bf16*  Ks   = Qs  + QS_ELEMS;             // [64][392] K
  bf16*  Vt   = Ks  + QS_ELEMS;             // [12][32][64] V transposed per head
  bf16*  Ps   = Vt  + VT_ELEMS;             // [64][72]  softmax probs (bf16)
  bf16*  OutS = Ps  + PS_ELEMS;             // [64][392] attention output
  float* Sc   = (float*)(OutS + QS_ELEMS);  // [64][68]  f32 scores
  int*   toff = (int*)(Sc + SC_ELEMS);      // [64] token -> r*56+c spatial offset

  const int tid  = threadIdx.x;
  const int lane = tid & 31;
  const int w    = tid >> 5;                // wave id 0..7
  const int half8 = (lane >> 4) << 3;       // 0 or 8: C/D row offset for hi lanes
  const int win  = blockIdx.x;              // 0..1023
  const int b    = win >> 6;
  const int wh   = (win >> 3) & 7;
  const int ww   = win & 7;
  const int sbase = (wh * 7) * 56 + ww * 7; // window origin in 56x56 plane
  const float* __restrict__ xb = x   + (size_t)b * DIM * 56 * 56;
  float*       __restrict__ ob = out + (size_t)b * DIM * 56 * 56;

  // ---- Phase 0: spatial offset table --------------------------------------
  if (tid < NP) toff[tid] = (tid / 7) * 56 + (tid % 7);
  __syncthreads();

  // ---- Phase 1: gather window from NCHW -> Xs bf16, zero-pad rows 49..63 --
  for (int i = tid; i < DIM * NP; i += 256) {
    int d = i >> 6;
    int t = i & 63;
    float v = 0.f;
    if (t < N_TOK) v = xb[(size_t)d * 3136 + sbase + toff[t]];
    Xs[t * XS_STRIDE + d] = (bf16)v;
  }
  __syncthreads();

  // ---- Phase 2: QKV GEMM  [64x384] @ [384x1152] + bqkv ---------------------
  // Waves split the 72 N-tiles; B fragments streamed from L2-resident wqkvT.
  // A 16-wide N tile never crosses a q/k/v third or a head -> uniform routing.
  for (int nt = w; nt < THREE_DIM / 16; nt += 8) {
    const int n0 = nt * 16;
    const bf16* btile = wqkvT + (size_t)n0 * DIM;
    if (nt + 8 < THREE_DIM / 16)
      __builtin_prefetch(wqkvT + (size_t)(n0 + 128) * DIM, 0, 1);
    v8f acc[4] = {};
#pragma unroll 1
    for (int kk = 0; kk < DIM / 32; ++kk) {
      v16bf bfrag = load_frag(btile, 0, kk * 32, DIM, lane);
#pragma unroll
      for (int mt = 0; mt < 4; ++mt) {
        v16bf afrag = load_frag(Xs, mt * 16, kk * 32, XS_STRIDE, lane);
        acc[mt] = wmma_bf16(afrag, bfrag, acc[mt]);
      }
    }
    const float bias = bqkv[n0 + (lane & 15)];
    const int which = n0 / DIM;              // uniform: 0=q 1=k 2=v
    const int np0   = n0 - which * DIM;      // uniform: 0..383
    if (which < 2) {
      // q/k: row-major [token][feat]; per-lane column -> strided b16 stores
      bf16* dst = (which == 0 ? Qs : Ks) + np0 + (lane & 15);
#pragma unroll
      for (int mt = 0; mt < 4; ++mt)
#pragma unroll
        for (int v = 0; v < 8; ++v)
          dst[(mt * 16 + v + half8) * QS_STRIDE] = (bf16)(acc[mt][v] + bias);
    } else {
      // v: transposed [head][dim][token]; 8 contiguous tokens -> one b128 store
      const int head = np0 >> 5;                       // uniform
      const int dim  = (np0 & 31) + (lane & 15);       // per-lane
      bf16* dst = Vt + (head * HD + dim) * NP;
#pragma unroll
      for (int mt = 0; mt < 4; ++mt) {
        v8bf pk;
#pragma unroll
        for (int v = 0; v < 8; ++v) pk[v] = (bf16)(acc[mt][v] + bias);
        *(v8bf*)(dst + mt * 16 + half8) = pk;
      }
    }
  }
  __syncthreads();

  // ---- Phase 3: per-head attention ----------------------------------------
  const float scale = 0.17677669529663687f;  // 32^-0.5
  for (int h = 0; h < HEADS; ++h) {
    // scores S = scale * Q_h @ K_h^T  (K_h row-major == Bt layout directly)
    for (int t = w; t < 16; t += 8) {
      int mt = t & 3, nt = t >> 2;
      v16bf afrag = load_frag(Qs, mt * 16, h * HD, QS_STRIDE, lane);
      v16bf bfrag = load_frag(Ks, nt * 16, h * HD, QS_STRIDE, lane);
      v8f acc = {};
      acc = wmma_bf16(afrag, bfrag, acc);
      int n = nt * 16 + (lane & 15);
#pragma unroll
      for (int v = 0; v < 8; ++v)
        Sc[(mt * 16 + v + half8) * S_STRIDE + n] = acc[v] * scale;
    }
    __syncthreads();

    // softmax: 4 threads per row, each owns a contiguous 16-column slab;
    // exponentials stay in registers, P written as 2x b128 per thread.
    {
      const int row = tid >> 2, sub = tid & 3;
      const float* srow = Sc + row * S_STRIDE + sub * 16;
      float e[16];
      float mx = -3.0e38f;
#pragma unroll
      for (int i = 0; i < 16; ++i) {
        int c = sub * 16 + i;
        float s = (c < N_TOK) ? srow[i] : -3.0e38f;
        e[i] = s;
        mx = fmaxf(mx, s);
      }
      mx = fmaxf(mx, __shfl_xor(mx, 1, 4));
      mx = fmaxf(mx, __shfl_xor(mx, 2, 4));
      float sum = 0.f;
#pragma unroll
      for (int i = 0; i < 16; ++i) {
        int c = sub * 16 + i;
        float ev = (c < N_TOK) ? __expf(e[i] - mx) : 0.f;
        e[i] = ev;
        sum += ev;
      }
      sum += __shfl_xor(sum, 1, 4);
      sum += __shfl_xor(sum, 2, 4);
      float inv = 1.0f / sum;
      v8bf p0, p1;
#pragma unroll
      for (int i = 0; i < 8; ++i) {
        p0[i] = (bf16)(e[i] * inv);
        p1[i] = (bf16)(e[8 + i] * inv);
      }
      bf16* prow = Ps + row * P_STRIDE + sub * 16;
      *(v8bf*)(prow)     = p0;
      *(v8bf*)(prow + 8) = p1;
    }
    __syncthreads();

    // O_h = P @ V_h : [64x64] @ [64x32]; Vt rows are head dims -> Bt layout
    {
      int mt = w & 3, nt = w >> 2;   // 8 waves cover 4x2 tiles
      v8f acc = {};
#pragma unroll
      for (int kk = 0; kk < 2; ++kk) {
        v16bf afrag = load_frag(Ps, mt * 16, kk * 32, P_STRIDE, lane);
        v16bf bfrag = load_frag(Vt + h * HD * NP, nt * 16, kk * 32, NP, lane);
        acc = wmma_bf16(afrag, bfrag, acc);
      }
      bf16* dst = OutS + h * HD + nt * 16 + (lane & 15);
#pragma unroll
      for (int v = 0; v < 8; ++v)
        dst[(mt * 16 + v + half8) * QS_STRIDE] = (bf16)acc[v];
    }
    __syncthreads();
  }

  // ---- Phase 4: proj GEMM [64x384]@[384x384] + bproj, scatter to NCHW -----
  for (int nt = w; nt < DIM / 16; nt += 8) {
    const int n0 = nt * 16;
    const bf16* btile = wprojT + (size_t)n0 * DIM;
    if (nt + 8 < DIM / 16)
      __builtin_prefetch(wprojT + (size_t)(n0 + 128) * DIM, 0, 1);
    v8f acc[4] = {};
#pragma unroll 1
    for (int kk = 0; kk < DIM / 32; ++kk) {
      v16bf bfrag = load_frag(btile, 0, kk * 32, DIM, lane);
#pragma unroll
      for (int mt = 0; mt < 4; ++mt) {
        v16bf afrag = load_frag(OutS, mt * 16, kk * 32, QS_STRIDE, lane);
        acc[mt] = wmma_bf16(afrag, bfrag, acc[mt]);
      }
    }
    const int ch     = n0 + (lane & 15);
    const float bias = bproj[ch];
    float* obc = ob + (size_t)ch * 3136 + sbase;
#pragma unroll
    for (int mt = 0; mt < 4; ++mt) {
#pragma unroll
      for (int v = 0; v < 8; ++v) {
        int m = mt * 16 + v + half8;
        if (m < N_TOK) obc[toff[m]] = acc[mt][v] + bias;
      }
    }
  }
}

extern "C" void kernel_launch(void* const* d_in, const int* in_sizes, int n_in,
                              void* d_out, int out_size, void* d_ws, size_t ws_size,
                              hipStream_t stream) {
  const float* x     = (const float*)d_in[0];
  const float* Wqkv  = (const float*)d_in[1];
  const float* bqkv  = (const float*)d_in[2];
  const float* Wproj = (const float*)d_in[3];
  const float* bproj = (const float*)d_in[4];
  float* out = (float*)d_out;

  // workspace: bf16 transposed weights (1,179,648 bytes total)
  bf16* wqkvT  = (bf16*)d_ws;
  bf16* wprojT = wqkvT + (size_t)DIM * THREE_DIM;

  const int prep_total = DIM * THREE_DIM + DIM * DIM;
  prep_weights<<<(prep_total + 255) / 256, 256, 0, stream>>>(Wqkv, Wproj, wqkvT, wprojT);

  // 1024 windows, one workgroup each, ~271KB dynamic LDS (<320KB/WGP)
  win_attn_fused<<<1024, 256, SMEM_BYTES, stream>>>(x, wqkvT, bqkv, wprojT, bproj, out);
}
